// SiameseNetwork_Debug_78658031059339
// MI455X (gfx1250) — matrix operand order repaired and checked by reading, not compiled, not run on hardware
//
#include <hip/hip_runtime.h>

// ---------------------------------------------------------------------------
// CDNA5 (gfx1250) wave32 WMMA implementation of the Siamese co-attention net.
// All GEMM-class work (WVa, both attention passes, both 3x3 convs) runs on
// v_wmma_f32_16x16x32_bf16.  Flash-attention streaming softmax avoids the
// 414MB S matrix entirely.
// ---------------------------------------------------------------------------

typedef __bf16 bf16_t;
typedef bf16_t bf16x8  __attribute__((ext_vector_type(8)));
typedef bf16_t bf16x16 __attribute__((ext_vector_type(16)));
typedef float  f32x8   __attribute__((ext_vector_type(8)));

#define NB   8      // batch
#define NC_  256    // channels C
#define NHW  3600   // 60*60
#define NH   60
#define K2C  4608   // 2C * 9 (implicit-GEMM K for 3x3 conv)
#define OUTW 480
#define NPIX (NB * NHW)   // 28800

__device__ __forceinline__ bf16x16 cat8(bf16x8 lo, bf16x8 hi) {
  return __builtin_shufflevector(lo, hi, 0,1,2,3,4,5,6,7,8,9,10,11,12,13,14,15);
}
// A fragment (16x32, M x K): lane = row (lane&15), K = 2*kbase + 0..15
// where kbase = (lane>>4)*8.  Source must be [M][K] rowmajor.
__device__ __forceinline__ bf16x16 frag_a(const bf16_t* row, int k0, int kbase) {
  bf16x8 lo = *(const bf16x8*)(row + k0 + kbase * 2);
  bf16x8 hi = *(const bf16x8*)(row + k0 + kbase * 2 + 8);
  return cat8(lo, hi);
}
// B fragment (32x16, K x N): lane = col (lane&15), K = lh*16 + 0..15
// (lh = lane>>4).  Source must be [N][K] rowmajor (K contiguous).
__device__ __forceinline__ bf16x16 frag_b(const bf16_t* col, int k0, int lh) {
  bf16x8 lo = *(const bf16x8*)(col + k0 + lh * 16);
  bf16x8 hi = *(const bf16x8*)(col + k0 + lh * 16 + 8);
  return cat8(lo, hi);
}
__device__ __forceinline__ f32x8 wmma_bf16(bf16x16 a, bf16x16 b, f32x8 c) {
  return __builtin_amdgcn_wmma_f32_16x16x32_bf16(false, a, false, b, (short)0, c,
                                                 false, false);
}
__device__ __forceinline__ f32x8 f32x8_zero() {
  f32x8 z = {0.f,0.f,0.f,0.f,0.f,0.f,0.f,0.f};
  return z;
}
__device__ __forceinline__ float sigmoidf_(float x) {
  return 1.f / (1.f + __expf(-x));
}

// ---------------------------------------------------------------------------
// Elementwise f32 -> bf16
__global__ void k_cvt1d(const float* __restrict__ src, bf16_t* __restrict__ dst, int n) {
  int i = blockIdx.x * blockDim.x + threadIdx.x;
  if (i < n) dst[i] = (bf16_t)src[i];
}

// f32 [B][C][HW]  ->  bf16 c-major copy [B][C][HW]  +  bf16 transpose [B][HW][C]
__global__ void k_cvt_tr(const float* __restrict__ src, bf16_t* __restrict__ dc,
                         bf16_t* __restrict__ dt) {
  int idx = blockIdx.x * blockDim.x + threadIdx.x;
  if (idx >= NB * NC_ * NHW) return;
  int i = idx % NHW;
  int c = (idx / NHW) % NC_;
  int b = idx / (NHW * NC_);
  bf16_t v = (bf16_t)src[idx];
  dc[idx] = v;
  dt[((size_t)b * NHW + i) * NC_ + c] = v;
}

// ---------------------------------------------------------------------------
// WVaT[b][i][d] = sum_c W_sim[d][c] * Va[b][c][i]
// A = W_sim [d][c] (K=c contiguous), B = VaT [i][c] (K=c contiguous).
__global__ __launch_bounds__(32)
void k_gemm_wva(const bf16_t* __restrict__ Wsim, const bf16_t* __restrict__ VaT,
                bf16_t* __restrict__ WVaT) {
  int b = blockIdx.z, d0 = blockIdx.y * 16, i0 = blockIdx.x * 16;
  int lane = threadIdx.x, lh = lane >> 4, lidx = lane & 15, kbase = lh * 8;
  const bf16_t* VaTb = VaT + (size_t)b * NHW * NC_;
  f32x8 acc = f32x8_zero();
#pragma unroll
  for (int kk = 0; kk < 8; ++kk) {
    int k0 = kk * 32;
    bf16x16 a  = frag_a(Wsim + (size_t)(d0 + lidx) * NC_, k0, kbase);
    bf16x16 bb = frag_b(VaTb + (size_t)(i0 + lidx) * NC_, k0, lh);
    acc = wmma_bf16(a, bb, acc);
  }
  bf16_t* out = WVaT + (size_t)b * NHW * NC_;
#pragma unroll
  for (int r = 0; r < 8; ++r) {
    int d = d0 + lh * 8 + r;      // D row
    int i = i0 + lidx;            // D col
    out[(size_t)i * NC_ + d] = (bf16_t)acc[r];
  }
}

// ---------------------------------------------------------------------------
// Flash attention pass.  out[b][c][pos_i] = sum_j softmax_j(Qt[i].Kt[j]) Vc[c][j]
//  Qt,Kt: [B][HW][C] bf16 (K=c contiguous)   Vc: [B][C][HW] bf16 (K=j contiguous)
// One wave per (b, 16-row i-tile); 16 f32x8 accumulators cover all 256 channels.
__global__ __launch_bounds__(32)
void k_attn(const bf16_t* __restrict__ Qt, const bf16_t* __restrict__ Kt,
            const bf16_t* __restrict__ Vc, float* __restrict__ Zout) {
  __shared__ __align__(16) bf16_t Qs[16][NC_];   // Q tile: 16 rows x 256 (8KB)
  __shared__ __align__(16) bf16_t Plds[16][32];  // P repack staging (1KB)
  const int b  = blockIdx.y;
  const int i0 = blockIdx.x * 16;
  const int lane = threadIdx.x, lh = lane >> 4, lidx = lane & 15, kbase = lh * 8;
  const bf16_t* Qb = Qt + (size_t)b * NHW * NC_;
  const bf16_t* Kb = Kt + (size_t)b * NHW * NC_;
  const bf16_t* Vb = Vc + (size_t)b * NC_ * NHW;

  { // cooperative Q tile load (rows are contiguous): 512 x 16B
    const uint4* s = (const uint4*)(Qb + (size_t)i0 * NC_);
    uint4* d = (uint4*)(&Qs[0][0]);
#pragma unroll
    for (int t = 0; t < 16; ++t) d[lane + t * 32] = s[lane + t * 32];
  }
  __syncthreads();

  f32x8 acc[16];
#pragma unroll
  for (int t = 0; t < 16; ++t) acc[t] = f32x8_zero();
  float mrow[8], lrow[8];
#pragma unroll
  for (int r = 0; r < 8; ++r) { mrow[r] = -1e30f; lrow[r] = 0.f; }

  for (int j0 = 0; j0 < NHW; j0 += 32) {
    if (j0 + 32 < NHW)
      __builtin_prefetch(Kb + (size_t)(j0 + 32 + lidx) * NC_, 0, 0);
    // ---- S tiles (16 x 32) ----
    f32x8 S0 = f32x8_zero(), S1 = f32x8_zero();
    int j0a = j0 + lidx;       if (j0a > NHW - 1) j0a = NHW - 1;
    int j1a = j0 + 16 + lidx;  if (j1a > NHW - 1) j1a = NHW - 1;
#pragma unroll
    for (int kk = 0; kk < 8; ++kk) {
      int k0 = kk * 32;
      bf16x16 a  = frag_a(&Qs[lidx][0], k0, kbase);
      bf16x16 b0 = frag_b(Kb + (size_t)j0a * NC_, k0, lh);
      bf16x16 b1 = frag_b(Kb + (size_t)j1a * NC_, k0, lh);
      S0 = wmma_bf16(a, b0, S0);
      S1 = wmma_bf16(a, b1, S1);
    }
    // mask invalid columns (only the tail step)
    if (j0 + lidx >= NHW) {
#pragma unroll
      for (int r = 0; r < 8; ++r) S0[r] = -1e30f;
    }
    if (j0 + 16 + lidx >= NHW) {
#pragma unroll
      for (int r = 0; r < 8; ++r) S1[r] = -1e30f;
    }
    // ---- streaming softmax stats (rows live on (lh, reg r)) ----
    float rm[8];
#pragma unroll
    for (int r = 0; r < 8; ++r) rm[r] = fmaxf(S0[r], S1[r]);
#pragma unroll
    for (int off = 1; off < 16; off <<= 1) {
#pragma unroll
      for (int r = 0; r < 8; ++r) rm[r] = fmaxf(rm[r], __shfl_xor(rm[r], off, 32));
    }
    float sc[8], rs[8];
#pragma unroll
    for (int r = 0; r < 8; ++r) {
      float mn = fmaxf(mrow[r], rm[r]);
      sc[r] = __expf(mrow[r] - mn);
      mrow[r] = mn;
      float p0 = __expf(S0[r] - mn);
      float p1 = __expf(S1[r] - mn);
      S0[r] = p0; S1[r] = p1;
      rs[r] = p0 + p1;
    }
#pragma unroll
    for (int off = 1; off < 16; off <<= 1) {
#pragma unroll
      for (int r = 0; r < 8; ++r) rs[r] += __shfl_xor(rs[r], off, 32);
    }
#pragma unroll
    for (int r = 0; r < 8; ++r) lrow[r] = lrow[r] * sc[r] + rs[r];
#pragma unroll
    for (int t = 0; t < 16; ++t) {
#pragma unroll
      for (int r = 0; r < 8; ++r) acc[t][r] *= sc[r];
    }
    // ---- repack P (D layout) -> A-fragment layout via LDS ----
#pragma unroll
    for (int r = 0; r < 8; ++r) {
      int row = lh * 8 + r;
      Plds[row][lidx]      = (bf16_t)S0[r];
      Plds[row][16 + lidx] = (bf16_t)S1[r];
    }
    __syncthreads();           // single-wave WG: S_NOP, ordering only
    bf16x16 pfrag = frag_a(&Plds[lidx][0], 0, kbase);
    // ---- P x V : accumulate all 256 channels ----
#pragma unroll
    for (int ct = 0; ct < 16; ++ct) {
      bf16x16 vfrag = frag_b(Vb + (size_t)(ct * 16 + lidx) * NHW + j0, 0, lh);
      acc[ct] = wmma_bf16(pfrag, vfrag, acc[ct]);
    }
    __syncthreads();
  }
  // ---- epilogue: divide by l, scatter to [B][C][HW] ----
#pragma unroll
  for (int ct = 0; ct < 16; ++ct) {
#pragma unroll
    for (int r = 0; r < 8; ++r) {
      int row = lh * 8 + r;           // i-local
      int cc  = ct * 16 + lidx;       // channel
      Zout[((size_t)b * NC_ + cc) * NHW + i0 + row] = acc[ct][r] / lrow[r];
    }
  }
}

// ---------------------------------------------------------------------------
// Gate: m = sigmoid(sum_c gw[c]*Z[b][c][i]);  Z[b][:][i] *= m
__global__ void k_gate(float* __restrict__ Z, const float* __restrict__ gw) {
  int idx = blockIdx.x * blockDim.x + threadIdx.x;
  if (idx >= NPIX) return;
  int b = idx / NHW, i = idx % NHW;
  float* Zb = Z + (size_t)b * NC_ * NHW + i;
  float s = 0.f;
  for (int c = 0; c < NC_; ++c) s += gw[c] * Zb[(size_t)c * NHW];
  float m = sigmoidf_(s);
  for (int c = 0; c < NC_; ++c) Zb[(size_t)c * NHW] *= m;
}

// ---------------------------------------------------------------------------
// 3x3 conv (concat[Z,V] -> C) as implicit GEMM, K = 2C*9 = 144*32.
// Each lane synthesizes its own B-fragment (fill pattern == B lane layout).
__global__ __launch_bounds__(32)
void k_conv3(const float* __restrict__ Z, const float* __restrict__ V,
             const bf16_t* __restrict__ Wc, float* __restrict__ Out) {
  int b = blockIdx.z, o0 = blockIdx.y * 16, p0 = blockIdx.x * 16;
  int lane = threadIdx.x, lh = lane >> 4, lidx = lane & 15, kbase = lh * 8;
  const float* Zb = Z + (size_t)b * NC_ * NHW;
  const float* Vb = V + (size_t)b * NC_ * NHW;
  int px = p0 + lidx, y = px / NH, x = px % NH;
  f32x8 acc = f32x8_zero();
  for (int kk = 0; kk < 144; ++kk) {
    int k0 = kk * 32;
    bf16x16 bb;
#pragma unroll
    for (int e = 0; e < 16; ++e) {
      int k = k0 + lh * 16 + e;
      int g = k / 9, rr = k % 9;
      int ky = rr / 3, kx = rr % 3;
      int y2 = y + ky - 1, x2 = x + kx - 1;
      float v = 0.f;
      if ((unsigned)y2 < (unsigned)NH && (unsigned)x2 < (unsigned)NH) {
        const float* src = (g < NC_) ? (Zb + (size_t)g * NHW)
                                     : (Vb + (size_t)(g - NC_) * NHW);
        v = src[y2 * NH + x2];
      }
      bb[e] = (bf16_t)v;
    }
    bf16x16 a = frag_a(Wc + (size_t)(o0 + lidx) * K2C, k0, kbase);
    acc = wmma_bf16(a, bb, acc);
  }
#pragma unroll
  for (int r = 0; r < 8; ++r) {
    int o = o0 + lh * 8 + r;
    Out[((size_t)b * NC_ + o) * NHW + p0 + lidx] = acc[r];
  }
}

// ---------------------------------------------------------------------------
// Training-mode BN stats -> per-channel affine (scale, shift)
__global__ __launch_bounds__(256)
void k_bn(const float* __restrict__ X, const float* __restrict__ gamma,
          const float* __restrict__ beta, float* __restrict__ scale,
          float* __restrict__ shift) {
  __shared__ float ssum[256], ssq[256];
  int o = blockIdx.x, t = threadIdx.x;
  float s = 0.f, q = 0.f;
  for (int n = t; n < NPIX; n += 256) {
    int b = n / NHW, i = n % NHW;
    float v = X[((size_t)b * NC_ + o) * NHW + i];
    s += v; q += v * v;
  }
  ssum[t] = s; ssq[t] = q;
  __syncthreads();
  for (int st = 128; st > 0; st >>= 1) {
    if (t < st) { ssum[t] += ssum[t + st]; ssq[t] += ssq[t + st]; }
    __syncthreads();
  }
  if (t == 0) {
    float mean = ssum[0] / (float)NPIX;
    float var  = ssq[0] / (float)NPIX - mean * mean;
    float sc   = gamma[o] * rsqrtf(var + 1e-5f);
    scale[o] = sc;
    shift[o] = beta[o] - mean * sc;
  }
}

// BN-apply + ReLU + 1x1 classifier (C -> 2)
__global__ void k_cls(const float* __restrict__ X, const float* __restrict__ scale,
                      const float* __restrict__ shift, const float* __restrict__ w,
                      const float* __restrict__ bias, float* __restrict__ out) {
  int idx = blockIdx.x * blockDim.x + threadIdx.x;
  if (idx >= NPIX) return;
  int b = idx / NHW, i = idx % NHW;
  float a0 = 0.f, a1 = 0.f;
  for (int o = 0; o < NC_; ++o) {
    float v = X[((size_t)b * NC_ + o) * NHW + i] * scale[o] + shift[o];
    v = fmaxf(v, 0.f);
    a0 += w[o] * v;
    a1 += w[NC_ + o] * v;
  }
  out[((size_t)b * 2 + 0) * NHW + i] = a0 + bias[0];
  out[((size_t)b * 2 + 1) * NHW + i] = a1 + bias[1];
}

// Bilinear 8x upsample (half-pixel centers, edge clamp) + sigmoid
__global__ void k_up(const float* __restrict__ X, float* __restrict__ out) {
  int idx = blockIdx.x * blockDim.x + threadIdx.x;
  const int total = NB * 2 * OUTW * OUTW;
  if (idx >= total) return;
  int x = idx % OUTW, y = (idx / OUTW) % OUTW, cc = idx / (OUTW * OUTW);
  float sy = (y + 0.5f) * (60.f / 480.f) - 0.5f;
  float sx = (x + 0.5f) * (60.f / 480.f) - 0.5f;
  int y0f = (int)floorf(sy); float wy = sy - (float)y0f;
  int x0f = (int)floorf(sx); float wx = sx - (float)x0f;
  int y0 = min(max(y0f, 0), 59), y1 = min(max(y0f + 1, 0), 59);
  int x0 = min(max(x0f, 0), 59), x1 = min(max(x0f + 1, 0), 59);
  const float* P = X + (size_t)cc * NHW;
  float v = (1.f - wy) * ((1.f - wx) * P[y0 * NH + x0] + wx * P[y0 * NH + x1]) +
            wy        * ((1.f - wx) * P[y1 * NH + x0] + wx * P[y1 * NH + x1]);
  out[idx] = sigmoidf_(v);
}

// ---------------------------------------------------------------------------
extern "C" void kernel_launch(void* const* d_in, const int* in_sizes, int n_in,
                              void* d_out, int out_size, void* d_ws, size_t ws_size,
                              hipStream_t stream) {
  const float* V_a    = (const float*)d_in[0];
  const float* V_b    = (const float*)d_in[1];
  const float* W_sim  = (const float*)d_in[2];
  const float* gate_w = (const float*)d_in[3];
  const float* cA_w   = (const float*)d_in[4];
  const float* cB_w   = (const float*)d_in[5];
  const float* bnA_g  = (const float*)d_in[6];
  const float* bnA_b  = (const float*)d_in[7];
  const float* bnB_g  = (const float*)d_in[8];
  const float* bnB_b  = (const float*)d_in[9];
  const float* clsA_w = (const float*)d_in[10];
  const float* clsA_b = (const float*)d_in[11];
  const float* clsB_w = (const float*)d_in[12];
  const float* clsB_b = (const float*)d_in[13];

  size_t off = 0;
  char* base = (char*)d_ws;
  auto alloc = [&](size_t bytes) -> void* {
    void* p = base + off;
    off += (bytes + 255) & ~(size_t)255;
    return p;
  };
  const size_t NBF = (size_t)NB * NC_ * NHW;      // 7,372,800 elements
  bf16_t* Wsim_bf = (bf16_t*)alloc(65536 * 2);
  bf16_t* Va_bf   = (bf16_t*)alloc(NBF * 2);
  bf16_t* Vb_bf   = (bf16_t*)alloc(NBF * 2);
  bf16_t* VaT     = (bf16_t*)alloc(NBF * 2);
  bf16_t* VbT     = (bf16_t*)alloc(NBF * 2);
  bf16_t* WVaT    = (bf16_t*)alloc(NBF * 2);
  bf16_t* WA_bf   = (bf16_t*)alloc((size_t)NC_ * K2C * 2);
  bf16_t* WB_bf   = (bf16_t*)alloc((size_t)NC_ * K2C * 2);
  float*  Za      = (float*)alloc(NBF * 4);
  float*  Zb      = (float*)alloc(NBF * 4);
  float*  convA   = (float*)alloc(NBF * 4);
  float*  convB   = (float*)alloc(NBF * 4);
  float*  bnAs    = (float*)alloc(NC_ * 4);
  float*  bnAh    = (float*)alloc(NC_ * 4);
  float*  bnBs    = (float*)alloc(NC_ * 4);
  float*  bnBh    = (float*)alloc(NC_ * 4);
  float*  x1      = (float*)alloc((size_t)NB * 2 * NHW * 4);
  float*  x2      = (float*)alloc((size_t)NB * 2 * NHW * 4);

  // 1) precision conversions / transposes
  k_cvt1d<<<(65536 + 255) / 256, 256, 0, stream>>>(W_sim, Wsim_bf, 65536);
  k_cvt1d<<<(NC_ * K2C + 255) / 256, 256, 0, stream>>>(cA_w, WA_bf, NC_ * K2C);
  k_cvt1d<<<(NC_ * K2C + 255) / 256, 256, 0, stream>>>(cB_w, WB_bf, NC_ * K2C);
  k_cvt_tr<<<(int)((NBF + 255) / 256), 256, 0, stream>>>(V_a, Va_bf, VaT);
  k_cvt_tr<<<(int)((NBF + 255) / 256), 256, 0, stream>>>(V_b, Vb_bf, VbT);

  // 2) WVaT = (W_sim @ Va)^T  [B][HW][C]
  k_gemm_wva<<<dim3(NHW / 16, NC_ / 16, NB), 32, 0, stream>>>(Wsim_bf, VaT, WVaT);

  // 3) flash attention, both directions
  //    Z_a[c,i] = sum_j softmax_j( WVa[:,i].Vb[:,j] ) Vb[c,j]
  k_attn<<<dim3(NHW / 16, NB), 32, 0, stream>>>(WVaT, VbT, Vb_bf, Za);
  //    Z_b[c,j] = sum_i softmax_i( WVa[:,i].Vb[:,j] ) Va[c,i]
  k_attn<<<dim3(NHW / 16, NB), 32, 0, stream>>>(VbT, WVaT, Va_bf, Zb);

  // 4) gating
  k_gate<<<(NPIX + 255) / 256, 256, 0, stream>>>(Za, gate_w);
  k_gate<<<(NPIX + 255) / 256, 256, 0, stream>>>(Zb, gate_w);

  // 5) 3x3 convs on concat[Z, V] (implicit GEMM, WMMA)
  k_conv3<<<dim3(NHW / 16, NC_ / 16, NB), 32, 0, stream>>>(Za, V_a, WA_bf, convA);
  k_conv3<<<dim3(NHW / 16, NC_ / 16, NB), 32, 0, stream>>>(Zb, V_b, WB_bf, convB);

  // 6) BN (training-mode batch stats) -> affine
  k_bn<<<NC_, 256, 0, stream>>>(convA, bnA_g, bnA_b, bnAs, bnAh);
  k_bn<<<NC_, 256, 0, stream>>>(convB, bnB_g, bnB_b, bnBs, bnBh);

  // 7) BN + ReLU + 1x1 classifier
  k_cls<<<(NPIX + 255) / 256, 256, 0, stream>>>(convA, bnAs, bnAh, clsA_w, clsA_b, x1);
  k_cls<<<(NPIX + 255) / 256, 256, 0, stream>>>(convB, bnBs, bnBh, clsB_w, clsB_b, x2);

  // 8) bilinear 8x upsample + sigmoid, stacked output [2][B][2][480][480]
  const int upN = NB * 2 * OUTW * OUTW;
  k_up<<<(upN + 255) / 256, 256, 0, stream>>>(x1, (float*)d_out);
  k_up<<<(upN + 255) / 256, 256, 0, stream>>>(x2, (float*)d_out + upN);
}
